// WindowAttention_kv_20366734917793
// MI455X (gfx1250) — matrix operand order repaired and checked by reading, not compiled, not run on hardware
//
#include <hip/hip_runtime.h>
#include <hip/hip_bf16.h>
#include <cstdint>

// ---------------- problem constants ----------------
#define WS_   7
#define NH_   12
#define DIM_  384
#define HD_   32
#define NTOK  343           // WS^3
#define NPAD  352           // 22 * 16
#define NT16  22            // NPAD / 16
#define BW    128           // windows * batch
#define NWIN  64            // mask windows
#define KVCOL 768           // 2*DIM
#define MROWS (BW * NTOK)   // 43904 = 686 * 64 exactly
#define SCALE_F 0.17677669529663687f   // 32^-0.5

typedef __attribute__((ext_vector_type(16))) __bf16 v16bf;
typedef __attribute__((ext_vector_type(8)))  float  v8f;
typedef __attribute__((ext_vector_type(4)))  unsigned int v4u;
typedef __attribute__((ext_vector_type(8)))  int v8i_;
typedef __attribute__((ext_vector_type(4)))  int v4i_;

union FragU {
    v16bf v;
    unsigned u[8];
};

__device__ __forceinline__ unsigned short f2bf(float f) {
    union { __bf16 h; unsigned short s; } cv;
    cv.h = (__bf16)f;
    return cv.s;
}

// A-fragment dword index within a 16-dword (32 bf16) row:
// lane<16 uses dwords {0..7}; lane>=16 uses {8..15}
__device__ __forceinline__ int a_dw(int j, int hi) {
    return j + (hi ? 8 : 0);
}

// LDS byte offset of a shared-memory address (low 32 bits of flat address)
__device__ __forceinline__ unsigned lds_off_of(const void* p) {
    return (unsigned)(size_t)p;
}

// CDNA5 async copy: 16 bytes global -> LDS per lane, tracked by ASYNCcnt
__device__ __forceinline__ void async_cp16(unsigned lds_off, const void* sbase, unsigned voff) {
    asm volatile("global_load_async_to_lds_b128 %0, %1, %2"
                 :: "v"(lds_off), "v"(voff), "s"(sbase) : "memory");
}
__device__ __forceinline__ void async_cp8(unsigned lds_off, const void* sbase, unsigned voff) {
    asm volatile("global_load_async_to_lds_b64 %0, %1, %2"
                 :: "v"(lds_off), "v"(voff), "s"(sbase) : "memory");
}
__device__ __forceinline__ void async_wait0() {
    asm volatile("s_wait_asynccnt 0" ::: "memory");
}
__device__ __forceinline__ void tensor_wait0() {
#if __has_builtin(__builtin_amdgcn_s_wait_tensorcnt)
    __builtin_amdgcn_s_wait_tensorcnt(0);
#else
    asm volatile("s_wait_tensorcnt 0" ::: "memory");
#endif
}

// ============================================================
// Kernel 1b: prep — build bf16 Q (scaled), zero pad rows of Q/K/V^T
// index space: ((b*12+h)*352 + n)*32 + d
// ============================================================
__global__ __launch_bounds__(256) void prep_kernel(
    const float* __restrict__ x_up,
    unsigned short* __restrict__ qb,
    unsigned short* __restrict__ kb,
    unsigned short* __restrict__ vbT)
{
    size_t idx = (size_t)blockIdx.x * 256 + threadIdx.x;
    const size_t total = (size_t)BW * NH_ * NPAD * HD_;
    if (idx >= total) return;
    int d  = (int)(idx & 31);
    int n  = (int)((idx >> 5) % NPAD);
    int bh = (int)((idx >> 5) / NPAD);
    if (n < NTOK) {
        int b = bh / NH_, h = bh % NH_;
        float q = x_up[((size_t)b * NTOK + n) * DIM_ + h * HD_ + d] * SCALE_F;
        qb[idx] = f2bf(q);
    } else {
        qb[idx] = 0;
        kb[idx] = 0;
        vbT[((size_t)bh * HD_ + d) * NPAD + n] = 0;
    }
}

// ============================================================
// Kernel 1: kv projection GEMM  (skip @ w_kv^T + b_kv) -> bf16 K, V^T
// grid (686, 12), block 256 (8 waves). 64x64 tile, K loop 12 x 32,
// double-buffered async-to-LDS staging overlapped with WMMA.
// ============================================================
__global__ __launch_bounds__(256) void kv_gemm_kernel(
    const float* __restrict__ skip,
    const float* __restrict__ w_kv,
    const float* __restrict__ b_kv,
    unsigned short* __restrict__ kb,
    unsigned short* __restrict__ vbT)
{
    __shared__ __align__(16) float As[2][64][32];
    __shared__ __align__(16) float Bs[2][64][32];

    const int mblk = blockIdx.x;
    const int nblk = blockIdx.y;
    const int t = threadIdx.x;
    const int wave = t >> 5, lane = t & 31;
    const int mt = wave & 3;          // which 16-row tile
    const int npair = wave >> 2;      // col tiles {2*npair, 2*npair+1}
    const int hi = lane >> 4;
    const int arow = lane & 15;

    const float* abase = skip + (size_t)mblk * 64 * DIM_;
    const float* bbase = w_kv + (size_t)nblk * 64 * DIM_;

    v8f c0 = {}; v8f c1 = {};

    // stage chunk kk into buffer `buf` (async, no wait)
    auto stage = [&](int buf, int kk) {
        unsigned al = lds_off_of(&As[buf][0][0]);
        unsigned bl = lds_off_of(&Bs[buf][0][0]);
        for (int i = t; i < 512; i += 256) {
            int r = i >> 3, c4 = (i & 7) * 4;
            unsigned voff = (unsigned)((r * DIM_ + kk + c4) * 4);
            async_cp16(al + i * 16, abase, voff);
            async_cp16(bl + i * 16, bbase, voff);
        }
    };

    stage(0, 0);
    async_wait0();
    __syncthreads();

    int buf = 0;
    for (int it = 0; it < DIM_ / 32; ++it) {
        if (it + 1 < DIM_ / 32) stage(buf ^ 1, (it + 1) * 32);

        FragU af;
        for (int j = 0; j < 8; ++j) {
            int dw = a_dw(j, hi);
            af.v[2*j]   = (__bf16)As[buf][mt*16 + arow][2*dw];
            af.v[2*j+1] = (__bf16)As[buf][mt*16 + arow][2*dw + 1];
        }
        FragU bf0, bf1;
        for (int j = 0; j < 8; ++j) {
            int k0 = 2*j + (hi ? 16 : 0);
            bf0.v[2*j]   = (__bf16)Bs[buf][npair*32 + arow][k0];
            bf0.v[2*j+1] = (__bf16)Bs[buf][npair*32 + arow][k0 + 1];
            bf1.v[2*j]   = (__bf16)Bs[buf][npair*32 + 16 + arow][k0];
            bf1.v[2*j+1] = (__bf16)Bs[buf][npair*32 + 16 + arow][k0 + 1];
        }
        c0 = __builtin_amdgcn_wmma_f32_16x16x32_bf16(false, af.v, false, bf0.v,
                                                     (short)0, c0, false, false);
        c1 = __builtin_amdgcn_wmma_f32_16x16x32_bf16(false, af.v, false, bf1.v,
                                                     (short)0, c1, false, false);
        async_wait0();
        __syncthreads();
        buf ^= 1;
    }

    // epilogue: +bias, scatter to kb (row-major [key][d]) / vbT ([d][key])
    const int col0 = nblk * 64 + npair * 32 + arow;
    const int col1 = col0 + 16;
    const int rbase = mblk * 64 + mt * 16 + (hi ? 8 : 0);
    for (int r = 0; r < 8; ++r) {
        int grow = rbase + r;
        int b = grow / NTOK, ntok = grow % NTOK;
        float v0 = c0[r] + b_kv[col0];
        float v1 = c1[r] + b_kv[col1];
        for (int which = 0; which < 2; ++which) {
            int col = which ? col1 : col0;
            float val = which ? v1 : v0;
            if (col < DIM_) {
                int h = col / HD_, d = col % HD_;
                kb[(((size_t)b * NH_ + h) * NPAD + ntok) * HD_ + d] = f2bf(val);
            } else {
                int cc = col - DIM_;
                int h = cc / HD_, d = cc % HD_;
                vbT[(((size_t)b * NH_ + h) * HD_ + d) * NPAD + ntok] = f2bf(val);
            }
        }
    }
}

// ============================================================
// Kernel 2: fused attention. grid (B*H = 1536, 22 q-tiles), block 128 (4 waves)
// K panel staged to LDS via Tensor Data Mover; Q tile via async copy.
// S = Q K^T (WMMA) -> +bias+mask -> softmax (unnormalized exp) -> P V (WMMA)
// ============================================================
__global__ __launch_bounds__(128) void attn_kernel(
    const unsigned short* __restrict__ qb,
    const unsigned short* __restrict__ kb,
    const unsigned short* __restrict__ vbT,
    const float* __restrict__ mask,
    const float* __restrict__ bias_table,
    const int* __restrict__ rpi,
    float* __restrict__ ob)
{
    __shared__ __align__(16) float          Sbuf[16][NPAD];    // 22.0 KB
    __shared__ __align__(16) unsigned short Pbuf[16][NPAD];    // 11.0 KB
    __shared__ __align__(16) unsigned short Ks[NPAD][HD_];     // 22.0 KB
    __shared__ __align__(16) unsigned short Qt[16][HD_];       //  1.0 KB
    __shared__ __align__(16) float          Opart[4][256];     //  4.0 KB
    __shared__ float rs_inv[16];

    const int bh = blockIdx.x;
    const int b = bh / NH_, h = bh % NH_;
    const int qbase = blockIdx.y * 16;
    const int t = threadIdx.x, wave = t >> 5, lane = t & 31;
    const int hi = lane >> 4;
    const int arow = lane & 15;

    const unsigned short* Kbase = kb + (size_t)bh * NPAD * HD_;

    // ---- TDM: stage the whole K panel (352x32 bf16 = 22528B) into LDS ----
#if __has_builtin(__builtin_amdgcn_tensor_load_to_lds)
    if (wave == 0) {
        const size_t ga = (size_t)(const void*)Kbase;
        v4u g0;
        g0.x = 1u;                                  // count=1 (valid descriptor)
        g0.y = lds_off_of(&Ks[0][0]);               // lds_addr
        g0.z = (unsigned)(ga & 0xffffffffu);        // global_addr[31:0]
        g0.w = (unsigned)((ga >> 32) & 0x01ffffffu) // global_addr[56:32]
             | 0x80000000u;                         // type=2 ("image")
        v8i_ g1;
        g1[0] = 0x00010000;          // workgroup_mask=0, data_size=1 (2 bytes)
        g1[1] = (int)(HD_ << 16);    // tensor_dim0[15:0]=32 at bits[63:48]
        g1[2] = (int)(NPAD << 16);   // dim0 hi=0 | tensor_dim1[15:0]=352
        g1[3] = (int)(HD_ << 16);    // dim1 hi=0 | tile_dim0=32
        g1[4] = NPAD;                // tile_dim1=352, tile_dim2=0
        g1[5] = HD_;                 // tensor_dim0_stride = 32 elements
        g1[6] = 0;
        g1[7] = 0;
        v4i_ z4 = {0, 0, 0, 0};
#if __clang_major__ >= 23
        v8i_ z8 = {0, 0, 0, 0, 0, 0, 0, 0};
        __builtin_amdgcn_tensor_load_to_lds(g0, g1, z4, z4, z8, 0);
#else
        __builtin_amdgcn_tensor_load_to_lds(g0, g1, z4, z4, 0);
#endif
        tensor_wait0();
    }
#else
    // fallback: cooperative async copy of K panel (1408 x 16B chunks)
    {
        unsigned kl = lds_off_of(&Ks[0][0]);
        for (int i = t; i < (NPAD * HD_ * 2) / 16; i += 128)
            async_cp16(kl + i * 16, Kbase, (unsigned)(i * 16));
        async_wait0();
    }
#endif

    // ---- stage Q tile (16x32 bf16 = 1KB) with CDNA5 async copy to LDS ----
    {
        const unsigned short* qsrc = qb + (((size_t)bh) * NPAD + qbase) * HD_;
        async_cp8(lds_off_of(&Qt[0][0]) + (unsigned)t * 8u, qsrc, (unsigned)t * 8u);
        async_wait0();
    }
    __syncthreads();

    // ---- A fragment (Q tile, shared by all waves) ----
    FragU aq;
    {
        const unsigned* qrow = (const unsigned*)&Qt[arow][0];
        for (int j = 0; j < 8; ++j) aq.u[j] = qrow[a_dw(j, hi)];
    }

    // ---- S = Q K^T : 22 key-tiles split over 4 waves, K from LDS ----
    for (int kt = wave; kt < NT16; kt += 4) {
        FragU bk;
        const unsigned* krow = (const unsigned*)&Ks[kt*16 + arow][0];
        for (int j = 0; j < 8; ++j) bk.u[j] = krow[(hi ? 8 : 0) + j];
        v8f s = {};
        s = __builtin_amdgcn_wmma_f32_16x16x32_bf16(false, aq.v, false, bk.v,
                                                    (short)0, s, false, false);
        for (int r = 0; r < 8; ++r)
            Sbuf[r + (hi ? 8 : 0)][kt * 16 + arow] = s[r];
    }
    __syncthreads();

    // ---- softmax: 8 threads per row, unnormalized exp into bf16 P ----
    {
        const int srow = t >> 3;
        const int sc = t & 7;
        const int qi = qbase + srow;
        const bool qvalid = qi < NTOK;
        const float* mrow = mask + (size_t)(b & (NWIN - 1)) * NTOK * NTOK + (size_t)qi * NTOK;
        const int* rrow = rpi + (size_t)qi * NTOK;
        float mx = -1e30f;
        for (int c = sc; c < NPAD; c += 8) {
            float v;
            if (qvalid && c < NTOK)
                v = Sbuf[srow][c] + bias_table[rrow[c] * NH_ + h] + mrow[c];
            else
                v = -1e30f;
            Sbuf[srow][c] = v;
            mx = fmaxf(mx, v);
        }
        for (int m = 1; m < 8; m <<= 1) mx = fmaxf(mx, __shfl_xor(mx, m, 8));
        float sum = 0.f;
        for (int c = sc; c < NPAD; c += 8) {
            float e = __expf(Sbuf[srow][c] - mx);
            sum += e;
            Pbuf[srow][c] = f2bf(e);
        }
        for (int m = 1; m < 8; m <<= 1) sum += __shfl_xor(sum, m, 8);
        if (sc == 0) rs_inv[srow] = 1.0f / sum;
    }
    __syncthreads();

    // ---- O = P V : 2 d-tiles x 11 k-strips, split over 4 waves ----
    const unsigned short* Vbase = vbT + (size_t)bh * HD_ * NPAD;
    const int tile = wave & 1;
    v8f o = {};
    for (int s = (wave >> 1); s < 11; s += 2) {
        const int dwb = s * 16;   // dword base of this 32-key strip
        FragU ap;
        const unsigned* prow = (const unsigned*)&Pbuf[arow][0];
        for (int j = 0; j < 8; ++j) ap.u[j] = prow[dwb + a_dw(j, hi)];
        FragU bv;
        const unsigned* vrow = (const unsigned*)(Vbase + (size_t)(tile*16 + arow) * NPAD);
        for (int j = 0; j < 8; ++j) bv.u[j] = vrow[dwb + (hi ? 8 : 0) + j];
        o = __builtin_amdgcn_wmma_f32_16x16x32_bf16(false, ap.v, false, bv.v,
                                                    (short)0, o, false, false);
    }
    for (int r = 0; r < 8; ++r) Opart[wave][lane * 8 + r] = o[r];
    __syncthreads();

    // ---- reduce partials (wave0+wave2 -> d-tile 0, wave1+wave3 -> d-tile 1) ----
    if (wave < 2) {
        for (int r = 0; r < 8; ++r) {
            float val = Opart[wave][lane * 8 + r] + Opart[wave + 2][lane * 8 + r];
            int row = r + (hi ? 8 : 0);
            int qi = qbase + row;
            if (qi < NTOK) {
                int d = wave * 16 + arow;
                ob[(((size_t)b * NTOK + qi) * NH_ + h) * HD_ + d] = val * rs_inv[row];
            }
        }
    }
}

// ============================================================
// Kernel 3: output projection GEMM  ((ob + pos_embed) @ w_proj^T + b_proj)
// grid (686, 6), block 256, double-buffered async staging.
// ============================================================
__global__ __launch_bounds__(256) void proj_gemm_kernel(
    const float* __restrict__ ob,
    const float* __restrict__ pos_embed,
    const float* __restrict__ w_proj,
    const float* __restrict__ b_proj,
    float* __restrict__ out)
{
    __shared__ __align__(16) float As[2][64][32];
    __shared__ __align__(16) float Ps[2][64][32];
    __shared__ __align__(16) float Bs[2][64][32];

    const int mblk = blockIdx.x;
    const int nblk = blockIdx.y;
    const int t = threadIdx.x;
    const int wave = t >> 5, lane = t & 31;
    const int mt = wave & 3;
    const int npair = wave >> 2;
    const int hi = lane >> 4;
    const int arow = lane & 15;

    const float* abase = ob + (size_t)mblk * 64 * DIM_;
    const float* pbase = pos_embed + (size_t)mblk * 64 * DIM_;
    const float* bbase = w_proj + (size_t)nblk * 64 * DIM_;

    v8f c0 = {}; v8f c1 = {};

    auto stage = [&](int buf, int kk) {
        unsigned al = lds_off_of(&As[buf][0][0]);
        unsigned pl = lds_off_of(&Ps[buf][0][0]);
        unsigned bl = lds_off_of(&Bs[buf][0][0]);
        for (int i = t; i < 512; i += 256) {
            int r = i >> 3, c4 = (i & 7) * 4;
            unsigned voff = (unsigned)((r * DIM_ + kk + c4) * 4);
            async_cp16(al + i * 16, abase, voff);
            async_cp16(pl + i * 16, pbase, voff);
            async_cp16(bl + i * 16, bbase, voff);
        }
    };

    stage(0, 0);
    async_wait0();
    __syncthreads();

    int buf = 0;
    for (int it = 0; it < DIM_ / 32; ++it) {
        if (it + 1 < DIM_ / 32) stage(buf ^ 1, (it + 1) * 32);

        FragU af;
        for (int j = 0; j < 8; ++j) {
            int dw = a_dw(j, hi);
            af.v[2*j]   = (__bf16)(As[buf][mt*16 + arow][2*dw] +
                                   Ps[buf][mt*16 + arow][2*dw]);
            af.v[2*j+1] = (__bf16)(As[buf][mt*16 + arow][2*dw + 1] +
                                   Ps[buf][mt*16 + arow][2*dw + 1]);
        }
        FragU bf0, bf1;
        for (int j = 0; j < 8; ++j) {
            int k0 = 2*j + (hi ? 16 : 0);
            bf0.v[2*j]   = (__bf16)Bs[buf][npair*32 + arow][k0];
            bf0.v[2*j+1] = (__bf16)Bs[buf][npair*32 + arow][k0 + 1];
            bf1.v[2*j]   = (__bf16)Bs[buf][npair*32 + 16 + arow][k0];
            bf1.v[2*j+1] = (__bf16)Bs[buf][npair*32 + 16 + arow][k0 + 1];
        }
        c0 = __builtin_amdgcn_wmma_f32_16x16x32_bf16(false, af.v, false, bf0.v,
                                                     (short)0, c0, false, false);
        c1 = __builtin_amdgcn_wmma_f32_16x16x32_bf16(false, af.v, false, bf1.v,
                                                     (short)0, c1, false, false);
        async_wait0();
        __syncthreads();
        buf ^= 1;
    }

    const int col0 = nblk * 64 + npair * 32 + arow;
    const int col1 = col0 + 16;
    const int rbase = mblk * 64 + mt * 16 + (hi ? 8 : 0);
    for (int r = 0; r < 8; ++r) {
        size_t grow = (size_t)rbase + r;
        out[grow * DIM_ + col0] = c0[r] + b_proj[col0];
        out[grow * DIM_ + col1] = c1[r] + b_proj[col1];
    }
}

// ============================================================
extern "C" void kernel_launch(void* const* d_in, const int* in_sizes, int n_in,
                              void* d_out, int out_size, void* d_ws, size_t ws_size,
                              hipStream_t stream) {
    const float* skip       = (const float*)d_in[0];
    const float* x_up       = (const float*)d_in[1];
    const float* pos_embed  = (const float*)d_in[2];
    const float* mask       = (const float*)d_in[3];
    const float* w_kv       = (const float*)d_in[4];
    const float* b_kv       = (const float*)d_in[5];
    const float* w_proj     = (const float*)d_in[6];
    const float* b_proj     = (const float*)d_in[7];
    const float* bias_table = (const float*)d_in[8];
    const int*   rpi        = (const int*)d_in[9];
    float* out = (float*)d_out;

    // workspace layout (bf16 Q/K/V^T padded to 352 tokens + f32 attn output)
    const size_t elems_bhnd = (size_t)BW * NH_ * NPAD * HD_;   // 17,301,504
    unsigned short* qb  = (unsigned short*)d_ws;
    unsigned short* kb  = qb + elems_bhnd;
    unsigned short* vbT = kb + elems_bhnd;
    float* ob = (float*)(vbT + elems_bhnd);
    // total ws use: 3*2*elems_bhnd + 4*BW*NTOK*DIM ~= 171 MB

    const size_t prep_total = elems_bhnd;
    prep_kernel<<<dim3((unsigned)((prep_total + 255) / 256)), dim3(256), 0, stream>>>(
        x_up, qb, kb, vbT);

    kv_gemm_kernel<<<dim3(MROWS / 64, KVCOL / 64), dim3(256), 0, stream>>>(
        skip, w_kv, b_kv, kb, vbT);

    attn_kernel<<<dim3(BW * NH_, NT16), dim3(128), 0, stream>>>(
        qb, kb, vbT, mask, bias_table, rpi, ob);

    proj_gemm_kernel<<<dim3(MROWS / 64, DIM_ / 64), dim3(256), 0, stream>>>(
        ob, pos_embed, w_proj, b_proj, out);
}
